// DeltaGRU_51522427683142
// MI455X (gfx1250) — matrix operand order repaired and checked by reading, not compiled, not run on hardware
//
#include <hip/hip_runtime.h>
#include <hip/hip_bf16.h>

typedef __attribute__((ext_vector_type(16))) _Float16 v16h;
typedef __attribute__((ext_vector_type(8)))  float    v8f;

#define HID   512
#define G3    1536
#define TSTEPS 1024
#define NWELEM (G3 * HID)   // 786432 elems per big weight matrix

__device__ __forceinline__ v8f splat8(float s) {
    v8f v;
#pragma unroll
    for (int i = 0; i < 8; ++i) v[i] = s;
    return v;
}

// sigmoid(x) = rcp(1 + exp(-x)); exp(-x)=inf -> rcp(inf)=0, safe.
__device__ __forceinline__ float sigmoid_f(float v) {
    return __builtin_amdgcn_rcpf(1.0f + __expf(-v));
}

// tanh(x) = (1 - e) * rcp(1 + e), e = exp(-2x); clamp to avoid inf*0.
__device__ __forceinline__ float tanh_f(float v) {
    float t = fminf(fmaxf(v, -20.0f), 20.0f);
    float e = __expf(-2.0f * t);
    return (1.0f - e) * __builtin_amdgcn_rcpf(1.0f + e);
}

__device__ __forceinline__ v8f wmma_f16(v16h a, v16h b, v8f c) {
    return __builtin_amdgcn_wmma_f32_16x16x32_f16(
        false, a, false, b, (short)0, c, false, false);
}

// Opaque zero in an SGPR: defeats LICM on loads whose address includes it,
// WITHOUT destroying the base pointer's global address-space provenance
// (keeps global_load, avoids flat_load + 64-bit vector address adds).
__device__ __forceinline__ int opaque_zero() {
    int v = 0;
    asm volatile("" : "+s"(v));
    return v;
}

// A fragment (16x32 f16, MxK) from LDS, row-major with given stride.
// Layout: lane m (0..15): K = 0..15 in v0..7;
// lane m+16: K = 16..31 (contiguous, matches B fragment).
__device__ __forceinline__ v16h load_a_frag(const _Float16* p, int stride,
                                            int kbase, int lane) {
    const int m = lane & 15;
    const int half = (lane >> 4) & 1;
    union { v16h v; unsigned u[8]; } r;
    const _Float16* row = p + m * stride + kbase + half * 16;
#pragma unroll
    for (int j = 0; j < 8; ++j)
        r.u[j] = *(const unsigned*)(row + 2 * j);
    return r.v;
}

// B fragment (32x16 f16, KxN) from global weight W[n][k] row-major.
// Lane L: column n = nbase + (L&15); lanes 0-15 hold K=kbase..+15,
// lanes 16-31 hold K=kbase+16..+31 (16 consecutive f16 = 32B contiguous).
__device__ __forceinline__ v16h load_b_frag(const _Float16* __restrict__ W,
                                            int stride, int nbase, int kbase,
                                            int lane) {
    const int n = nbase + (lane & 15);
    const int k = kbase + ((lane >> 4) & 1) * 16;
    return *(const v16h*)(W + (size_t)n * stride + k);
}

// ---------------- prep: f32 -> f16 weights, pad W_ih_l0 K 6->32 ------------
__global__ void prep_weights(const float* __restrict__ Wih0,
                             const float* __restrict__ Whh0,
                             const float* __restrict__ Wih1,
                             const float* __restrict__ Whh1,
                             _Float16* __restrict__ ws) {
    int i = blockIdx.x * 256 + threadIdx.x;
    if (i < NWELEM) {
        ws[i]              = (_Float16)Whh0[i];
        ws[NWELEM + i]     = (_Float16)Wih1[i];
        ws[2 * NWELEM + i] = (_Float16)Whh1[i];
    }
    if (i < G3 * 32) {
        int n = i >> 5, k = i & 31;
        ws[3 * NWELEM + i] = (_Float16)(k < 6 ? Wih0[n * 6 + k] : 0.0f);
    }
}

// ---------------- main persistent DeltaGRU kernel --------------------------
// Grid: 4 blocks x 512 threads (16 waves). Block g handles batch rows
// [16g, 16g+16). Wave w owns hidden units [32w, 32w+32) for both layers.
__global__ __launch_bounds__(512) void delta_gru_kernel(
    const float* __restrict__ x,
    const float* __restrict__ bih0, const float* __restrict__ bhh0,
    const float* __restrict__ bih1, const float* __restrict__ bhh1,
    const float* __restrict__ Wfc,  const float* __restrict__ bfc,
    const _Float16* __restrict__ Whh0h, const _Float16* __restrict__ Wih1h,
    const _Float16* __restrict__ Whh1h, const _Float16* __restrict__ Wih0p,
    float* __restrict__ out)
{
    __shared__ _Float16 s_dh0[16 * HID];
    __shared__ _Float16 s_dx1[16 * HID];
    __shared__ _Float16 s_dh1[16 * HID];
    __shared__ _Float16 s_dx0[16 * 32];
    __shared__ float    s_out[16 * 2];

    const int tid  = threadIdx.x;
    const int lane = tid & 31;
    const int wave = tid >> 5;            // 0..15
    const int wg   = blockIdx.x;
    const int half = (lane >> 4) & 1;
    const int col  = lane & 15;

    for (int i = tid; i < 16 * HID; i += 512) {
        s_dh0[i] = (_Float16)0.0f;
        s_dx1[i] = (_Float16)0.0f;
        s_dh1[i] = (_Float16)0.0f;
    }
    if (tid < 16 * 32) s_dx0[tid] = (_Float16)0.0f;
    if (tid < 32) s_out[tid] = 0.0f;

    // Persistent per-wave state (2 hidden tiles per wave).
    v8f dmr0[2], dmz0[2], dmn0[2], dmnh0[2], h0[2], hp0[2], x1p[2];
    v8f dmr1[2], dmz1[2], dmn1[2], dmnh1[2], h1[2], hp1[2];
    float wf0[2], wf1[2];
#pragma unroll
    for (int u = 0; u < 2; ++u) {
        const int j = (wave * 2 + u) * 16 + col;       // hidden unit index
        dmr0[u]  = splat8(bih0[j] + bhh0[j]);
        dmz0[u]  = splat8(bih0[HID + j] + bhh0[HID + j]);
        dmn0[u]  = splat8(bih0[2 * HID + j]);
        dmnh0[u] = splat8(bhh0[2 * HID + j]);
        dmr1[u]  = splat8(bih1[j] + bhh1[j]);
        dmz1[u]  = splat8(bih1[HID + j] + bhh1[HID + j]);
        dmn1[u]  = splat8(bih1[2 * HID + j]);
        dmnh1[u] = splat8(bhh1[2 * HID + j]);
        h0[u] = splat8(0.f); hp0[u] = splat8(0.f); x1p[u] = splat8(0.f);
        h1[u] = splat8(0.f); hp1[u] = splat8(0.f);
        wf0[u] = Wfc[j];
        wf1[u] = Wfc[HID + j];
    }
    float xp[6] = {0.f, 0.f, 0.f, 0.f, 0.f, 0.f};   // wave0 feature memory
    const float bfc0 = bfc[0], bfc1 = bfc[1];

    for (int t = 0; t < TSTEPS; ++t) {
        __syncthreads();                                  // fence: out reset / dx0 reuse
        // ---- Phase A: feature expansion + input delta (wave0, 16 lanes) --
        if (wave == 0 && lane < 16) {
            const int b = wg * 16 + lane;
            const float iv = x[((size_t)b * TSTEPS + t) * 2 + 0];
            const float qv = x[((size_t)b * TSTEPS + t) * 2 + 1];
            const float amp = sqrtf(iv * iv + qv * qv);
            float f[6] = {iv, qv, amp, amp * amp * amp, qv / amp, iv / amp};
#pragma unroll
            for (int k = 0; k < 6; ++k) {
                float d  = f[k] - xp[k];
                float ad = fabsf(d);
                float dq = (ad < 0.1f) ? 0.0f : d;
                if (ad >= 0.1f) xp[k] = f[k];
                s_dx0[lane * 32 + k] = (_Float16)dq;
            }
        }
        __syncthreads();                                  // (1)

        // ---- Phase B: layer0 GEMMs --------------------------------------
        {
            // Opaque zero offset: per-step fresh L2 loads of the six input-
            // weight fragments; base pointer provenance kept -> global_load.
            const _Float16* wih = Wih0p + opaque_zero();
            v16h a0 = load_a_frag(s_dx0, 32, 0, lane);
#pragma unroll
            for (int u = 0; u < 2; ++u) {
                const int nb = (wave * 2 + u) * 16;
                dmr0[u] = wmma_f16(a0, load_b_frag(wih, 32, nb,            0, lane), dmr0[u]);
                dmz0[u] = wmma_f16(a0, load_b_frag(wih, 32, HID + nb,     0, lane), dmz0[u]);
                dmn0[u] = wmma_f16(a0, load_b_frag(wih, 32, 2 * HID + nb, 0, lane), dmn0[u]);
            }
            for (int kt = 0; kt < HID; kt += 32) {
                v16h ah = load_a_frag(s_dh0, HID, kt, lane);
#pragma unroll
                for (int u = 0; u < 2; ++u) {
                    const int nb = (wave * 2 + u) * 16;
                    dmr0[u]  = wmma_f16(ah, load_b_frag(Whh0h, HID, nb,            kt, lane), dmr0[u]);
                    dmz0[u]  = wmma_f16(ah, load_b_frag(Whh0h, HID, HID + nb,     kt, lane), dmz0[u]);
                    dmnh0[u] = wmma_f16(ah, load_b_frag(Whh0h, HID, 2 * HID + nb, kt, lane), dmnh0[u]);
                }
            }
        }
        __syncthreads();                                  // (2) done reading s_dh0

        // ---- Layer0 pointwise: gates, h update, deltas, LDS publish ------
#pragma unroll
        for (int u = 0; u < 2; ++u) {
            const int j = (wave * 2 + u) * 16 + col;
#pragma unroll
            for (int r = 0; r < 8; ++r) {
                const float rr = sigmoid_f(dmr0[u][r]);
                const float zz = sigmoid_f(dmz0[u][r]);
                const float nn = tanh_f(dmn0[u][r] + rr * dmnh0[u][r]);
                const float hn = (1.0f - zz) * nn + zz * h0[u][r];
                h0[u][r] = hn;
                float dh  = hn - hp0[u][r];
                float adh = fabsf(dh);
                float dhq = (adh < 0.05f) ? 0.0f : dh;
                hp0[u][r] = (adh >= 0.05f) ? hn : hp0[u][r];
                float dx  = hn - x1p[u][r];
                float adx = fabsf(dx);
                float dxq = (adx < 0.1f) ? 0.0f : dx;
                x1p[u][r] = (adx >= 0.1f) ? hn : x1p[u][r];
                const int m = r + 8 * half;
                s_dh0[m * HID + j] = (_Float16)dhq;
                s_dx1[m * HID + j] = (_Float16)dxq;
            }
        }
        __syncthreads();                                  // (3)

        // ---- Phase C: layer1 GEMMs --------------------------------------
        for (int kt = 0; kt < HID; kt += 32) {
            v16h axx = load_a_frag(s_dx1, HID, kt, lane);
            v16h ahh = load_a_frag(s_dh1, HID, kt, lane);
#pragma unroll
            for (int u = 0; u < 2; ++u) {
                const int nb = (wave * 2 + u) * 16;
                dmr1[u]  = wmma_f16(axx, load_b_frag(Wih1h, HID, nb,            kt, lane), dmr1[u]);
                dmz1[u]  = wmma_f16(axx, load_b_frag(Wih1h, HID, HID + nb,     kt, lane), dmz1[u]);
                dmn1[u]  = wmma_f16(axx, load_b_frag(Wih1h, HID, 2 * HID + nb, kt, lane), dmn1[u]);
                dmr1[u]  = wmma_f16(ahh, load_b_frag(Whh1h, HID, nb,            kt, lane), dmr1[u]);
                dmz1[u]  = wmma_f16(ahh, load_b_frag(Whh1h, HID, HID + nb,     kt, lane), dmz1[u]);
                dmnh1[u] = wmma_f16(ahh, load_b_frag(Whh1h, HID, 2 * HID + nb, kt, lane), dmnh1[u]);
            }
        }
        __syncthreads();                                  // (4) done reading s_dh1/s_dx1

        // ---- Layer1 pointwise + FC partials -----------------------------
        float po0[8], po1[8];
#pragma unroll
        for (int r = 0; r < 8; ++r) { po0[r] = 0.f; po1[r] = 0.f; }
#pragma unroll
        for (int u = 0; u < 2; ++u) {
            const int j = (wave * 2 + u) * 16 + col;
#pragma unroll
            for (int r = 0; r < 8; ++r) {
                const float rr = sigmoid_f(dmr1[u][r]);
                const float zz = sigmoid_f(dmz1[u][r]);
                const float nn = tanh_f(dmn1[u][r] + rr * dmnh1[u][r]);
                const float hn = (1.0f - zz) * nn + zz * h1[u][r];
                h1[u][r] = hn;
                float dh  = hn - hp1[u][r];
                float adh = fabsf(dh);
                float dhq = (adh < 0.05f) ? 0.0f : dh;
                hp1[u][r] = (adh >= 0.05f) ? hn : hp1[u][r];
                const int m = r + 8 * half;
                s_dh1[m * HID + j] = (_Float16)dhq;
                po0[r] += hn * wf0[u];
                po1[r] += hn * wf1[u];
            }
        }
        // Reduce FC partials over the 16 lanes sharing the same batch rows.
#pragma unroll
        for (int r = 0; r < 8; ++r) {
            float v0 = po0[r], v1 = po1[r];
#pragma unroll
            for (int mask = 1; mask < 16; mask <<= 1) {
                v0 += __shfl_xor(v0, mask, 32);
                v1 += __shfl_xor(v1, mask, 32);
            }
            if (col == 0) {
                const int m = r + 8 * half;
                atomicAdd(&s_out[m * 2 + 0], v0);
                atomicAdd(&s_out[m * 2 + 1], v1);
            }
        }
        __syncthreads();                                  // (5)
        if (tid < 32) {
            const int m = tid >> 1, o = tid & 1;
            const int b = wg * 16 + m;
            out[((size_t)b * TSTEPS + t) * 2 + o] =
                s_out[tid] + (o ? bfc1 : bfc0);
            s_out[tid] = 0.0f;
        }
    }
}

extern "C" void kernel_launch(void* const* d_in, const int* in_sizes, int n_in,
                              void* d_out, int out_size, void* d_ws, size_t ws_size,
                              hipStream_t stream) {
    (void)in_sizes; (void)n_in; (void)out_size; (void)ws_size;
    const float* x    = (const float*)d_in[0];
    const float* Wih0 = (const float*)d_in[2];
    const float* Whh0 = (const float*)d_in[3];
    const float* bih0 = (const float*)d_in[4];
    const float* bhh0 = (const float*)d_in[5];
    const float* Wih1 = (const float*)d_in[6];
    const float* Whh1 = (const float*)d_in[7];
    const float* bih1 = (const float*)d_in[8];
    const float* bhh1 = (const float*)d_in[9];
    const float* Wfc  = (const float*)d_in[10];
    const float* bfc  = (const float*)d_in[11];
    _Float16* ws = (_Float16*)d_ws;

    prep_weights<<<(NWELEM + 255) / 256, 256, 0, stream>>>(Wih0, Whh0, Wih1, Whh1, ws);
    delta_gru_kernel<<<4, 512, 0, stream>>>(
        x, bih0, bhh0, bih1, bhh1, Wfc, bfc,
        ws, ws + NWELEM, ws + 2 * NWELEM, ws + 3 * NWELEM,
        (float*)d_out);
}
